// GNN_PyG_60327110639967
// MI455X (gfx1250) — matrix operand order repaired and checked by reading, not compiled, not run on hardware
//
#include <hip/hip_runtime.h>

// ---------------------------------------------------------------------------
// MI455X / gfx1250, wave32. Dense layers via v_wmma_f32_16x16x32_f16
// (f16 A/B, f32 accum). K/N/BIAS/RELU are template constants; the full-tile
// vs partial-tile store path is chosen once per wave (uniform branch), so the
// hot loop is straight-line: 16 strided 16-bit LDS loads -> v_wmma -> 8 immediate-offset
// stores. Edge-stream tensors (~205 MB) use non-temporal policy so the
// 192 MB L2 keeps node tensors resident for the random per-edge gathers.
// ---------------------------------------------------------------------------

typedef __attribute__((ext_vector_type(16))) _Float16 v16h;
typedef __attribute__((ext_vector_type(8)))  float    v8f;
typedef __attribute__((ext_vector_type(4)))  float    v4f;

#define NEG_SLOPE 0.2f

template<bool NT> __device__ inline v4f ld4(const float* p) {
    if constexpr (NT) return __builtin_nontemporal_load((const v4f*)p);
    else              return *(const v4f*)p;
}
template<bool NT> __device__ inline void st1(float* p, float v) {
    if constexpr (NT) __builtin_nontemporal_store(v, p);
    else              *p = v;
}

// ---- tile loop (compute + store), FULL selects the guard-free store path ----
template<int N, int NPAD, int nK, bool NTY, bool BIAS, bool RELU, bool FULL>
__device__ inline void gemm_tiles(const v16h* a, const _Float16* sW,
                                  const float* __restrict__ bias,
                                  float* __restrict__ Y,
                                  int M, int rowBase, int lane, int hi8)
{
    constexpr int nT = NPAD / 16;
#pragma unroll
    for (int t = 0; t < nT; ++t) {
        v8f c = {0.f, 0.f, 0.f, 0.f, 0.f, 0.f, 0.f, 0.f};
#pragma unroll
        for (int s = 0; s < nK; ++s) {
            // B layout: lane l holds column (l&15); halves cover k = kb..kb+7, kb+16..kb+23, kb = s*32 + hi8
            v16h b;
#pragma unroll
            for (int j = 0; j < 16; ++j)
                b[j] = sW[(s * 32 + hi8 + (j < 8 ? j : j + 8)) * NPAD + t * 16 + (lane & 15)];
            c = __builtin_amdgcn_wmma_f32_16x16x32_f16(
                    false, a[s], false, b, (short)0, c, false, false);
        }
        const int col = t * 16 + (lane & 15);
        if (col < N) {                              // folds away when N==NPAD
            float bv = 0.0f;
            if constexpr (BIAS) bv = bias[col];
            float* yp = Y + (size_t)(rowBase + hi8) * N + col;
#pragma unroll
            for (int i = 0; i < 8; ++i) {           // C: VGPR i -> M = hi8 + i
                if (FULL || (rowBase + hi8 + i < M)) {
                    float v = c[i] + bv;
                    if constexpr (RELU) v = fmaxf(v, 0.0f);
                    st1<NTY>(yp + i * N, v);        // immediate offsets
                }
            }
        }
    }
}

// ------------------------- generic WMMA linear layer ------------------------
// Y[M,N] = act(X[M,K] @ W[K,N] + bias). Block = 256 threads = 8 waves; each
// wave owns 16 rows and covers N in 16-wide WMMA tiles (fully unrolled).
// W staged once per block into LDS as f16 (zero-padded to KP x NPAD).
template<int K, int N, bool NTX, bool NTY, bool BIAS, bool RELU>
__global__ __launch_bounds__(256) void wmma_linear_t(
    const float* __restrict__ X, const float* __restrict__ W,
    const float* __restrict__ bias, float* __restrict__ Y, int M)
{
    constexpr int KP   = (K + 31) & ~31;
    constexpr int NPAD = (N + 15) & ~15;
    constexpr int nK   = KP / 32;
    __shared__ alignas(32) _Float16 sW[KP * NPAD];

    // cooperative load W -> LDS (f32 -> f16), zero pad (guards fold when exact)
    for (int idx = threadIdx.x; idx < KP * NPAD; idx += 256) {
        int k = idx / NPAD, n = idx % NPAD;
        float v = (k < K && n < N) ? W[k * N + n] : 0.0f;
        sW[idx] = (_Float16)v;
    }
    __syncthreads();

    const int waveId  = threadIdx.x >> 5;
    const int lane    = threadIdx.x & 31;
    const int rowBase = blockIdx.x * 128 + waveId * 16;
    if (rowBase >= M) return;               // wave-uniform: EXEC stays full

    int row = rowBase + (lane & 15);
    if (row >= M) row = M - 1;              // clamp: loads stay in-bounds
    const size_t xoff = (size_t)row * (size_t)K;
    const int hi8 = (lane >> 4) << 3;       // 0 or 8

    // A fragments, 16-bit 16x32 WMMA layout. Lane l holds row (l&15); its 16
    // halves cover k = kb..kb+7 and kb+16..kb+23, kb = s*32 + hi8 -> two runs
    // of 8 consecutive k: 4x b128 loads. Group validity folds at compile time.
    v16h a[nK];
    const v4f z4 = {0.f, 0.f, 0.f, 0.f};
#pragma unroll
    for (int s = 0; s < nK; ++s) {
        const int kb = s * 32 + hi8;
        const bool g0 = (s * 32 + 8  <= K);   // lane-uniform, constant
        const bool g1 = (s * 32 + 24 <= K);
        v4f f0 = g0 ? ld4<NTX>(X + xoff + kb)      : z4;
        v4f f1 = g0 ? ld4<NTX>(X + xoff + kb + 4)  : z4;
        v4f f2 = g1 ? ld4<NTX>(X + xoff + kb + 16) : z4;
        v4f f3 = g1 ? ld4<NTX>(X + xoff + kb + 20) : z4;
#pragma unroll
        for (int j = 0; j < 4; ++j) {
            a[s][j]      = (_Float16)f0[j];
            a[s][4 + j]  = (_Float16)f1[j];
            a[s][8 + j]  = (_Float16)f2[j];
            a[s][12 + j] = (_Float16)f3[j];
        }
    }

    // single wave-uniform branch: straight-line stores in the common case
    if (rowBase + 16 <= M)
        gemm_tiles<N, NPAD, nK, NTY, BIAS, RELU, true >(a, sW, bias, Y, M, rowBase, lane, hi8);
    else
        gemm_tiles<N, NPAD, nK, NTY, BIAS, RELU, false>(a, sW, bias, Y, M, rowBase, lane, hi8);
}

template<int K, int N, bool NTX, bool NTY, bool BIAS, bool RELU>
static void L(const float* X, const float* W, const float* bias, float* Y,
              int M, hipStream_t s)
{
    wmma_linear_t<K, N, NTX, NTY, BIAS, RELU><<<(M + 127) / 128, 256, 0, s>>>(X, W, bias, Y, M);
}

// ------------------------- helpers --------------------------------------
__device__ inline unsigned flipF(float f) {
    unsigned u = __float_as_uint(f);
    return (u & 0x80000000u) ? ~u : (u | 0x80000000u);
}
__device__ inline float unflipF(unsigned u) {
    unsigned v = (u & 0x80000000u) ? (u & 0x7FFFFFFFu) : ~u;
    return __uint_as_float(v);
}

__global__ void fill_u32(unsigned* p, unsigned v, size_t n) {
    size_t i = (size_t)blockIdx.x * blockDim.x + threadIdx.x;
    if (i < n) p[i] = v;
}

__global__ void init_rows(float* out, const float* __restrict__ bias,
                          size_t total, int Dmask) {   // D is a power of two
    size_t i = (size_t)blockIdx.x * blockDim.x + threadIdx.x;
    if (i < total) out[i] = bias[i & Dmask];
}

__global__ void relu_k(float* p, size_t n) {
    size_t i = (size_t)blockIdx.x * blockDim.x + threadIdx.x;
    if (i < n) p[i] = fmaxf(p[i], 0.0f);
}

// ------------------------- GATv2 attention passes ------------------------
// Pass 1: wave per edge: logit = dot(leaky(xl[src]+xr[dst]+ea[e]), att);
// segment max over dst via ordered-uint atomicMax (node rows L2-resident,
// edge stream ea read non-temporally).
template<int D>
__global__ void gat_logits(const float* __restrict__ xl, const float* __restrict__ xr,
                           const float* __restrict__ ea, const int* __restrict__ ei,
                           const float* __restrict__ att,
                           float* __restrict__ logitsB, unsigned* __restrict__ nmax,
                           int nE)
{
    int wid  = (int)((blockIdx.x * (size_t)blockDim.x + threadIdx.x) >> 5);
    int lane = threadIdx.x & 31;
    if (wid >= nE) return;
    int src = ei[wid], dst = ei[nE + wid];
    float part = 0.f;
    int d = lane * 4;
    if (d < D) {
        v4f a4 = ld4<false>(&xl[(size_t)src * D + d]);
        v4f b4 = ld4<false>(&xr[(size_t)dst * D + d]);
        v4f e4 = ld4<true >(&ea[(size_t)wid * D + d]);
        v4f t4 = ld4<false>(&att[d]);
#pragma unroll
        for (int j = 0; j < 4; ++j) {
            float m = a4[j] + b4[j] + e4[j];
            part += (m > 0.f ? m : NEG_SLOPE * m) * t4[j];
        }
    }
    for (int off = 16; off; off >>= 1) part += __shfl_xor(part, off, 32);
    if (lane == 0) {
        logitsB[wid] = part;
        atomicMax(&nmax[dst], flipF(part));
    }
}

// Pass 2: thread per edge: ex = exp(logit - max[dst]); den[dst] += ex
__global__ void gat_expden(float* logitsB, const unsigned* __restrict__ nmax,
                           float* den, const int* __restrict__ ei, int nE)
{
    int e = blockIdx.x * blockDim.x + threadIdx.x;
    if (e >= nE) return;
    int dst = ei[nE + e];
    float ex = __expf(logitsB[e] - unflipF(nmax[dst]));
    logitsB[e] = ex;
    atomicAdd(&den[dst], ex);
}

// Pass 3: wave per edge: out[dst] += xl[src] * (ex / den[dst])
template<int D>
__global__ void gat_scatter(const float* __restrict__ xl, const float* __restrict__ logitsB,
                            const float* __restrict__ den, const int* __restrict__ ei,
                            float* out, int nE)
{
    int wid  = (int)((blockIdx.x * (size_t)blockDim.x + threadIdx.x) >> 5);
    int lane = threadIdx.x & 31;
    if (wid >= nE) return;
    int src = ei[wid], dst = ei[nE + wid];
    float alpha = logitsB[wid] / den[dst];
    int d = lane * 4;
    if (d < D) {
        v4f v = ld4<false>(&xl[(size_t)src * D + d]);
        float* o = &out[(size_t)dst * D + d];
        atomicAdd(o + 0, v[0] * alpha);
        atomicAdd(o + 1, v[1] * alpha);
        atomicAdd(o + 2, v[2] * alpha);
        atomicAdd(o + 3, v[3] * alpha);
    }
}

// ------------------------- pooling + critic head --------------------------
__global__ void pool_sum(const float* __restrict__ h, const int* __restrict__ batch,
                         float* pooled, int nN)
{
    size_t i = (size_t)blockIdx.x * blockDim.x + threadIdx.x;
    if (i >= (size_t)nN * 128) return;
    int n = (int)(i >> 7), d = (int)(i & 127);
    atomicAdd(&pooled[batch[n] * 128 + d], h[i]);
}

__global__ void pool_cnt(const int* __restrict__ batch, float* counts, int nN)
{
    int n = blockIdx.x * blockDim.x + threadIdx.x;
    if (n < nN) atomicAdd(&counts[batch[n]], 1.0f);
}

__global__ void graph_values(const float* __restrict__ pooled, const float* __restrict__ counts,
                             const float* __restrict__ Wc, const float* __restrict__ bc,
                             float* values, int G)
{
    int g = blockIdx.x, lane = threadIdx.x;
    if (g >= G) return;
    float s = 0.f;
    for (int d = lane; d < 128; d += 32) s += pooled[g * 128 + d] * Wc[d];
    for (int off = 16; off; off >>= 1) s += __shfl_xor(s, off, 32);
    if (lane == 0) {
        float c = fmaxf(counts[g], 1.0f);
        values[g] = s / c + bc[0];
    }
}

// ------------------------- host-side round helper -------------------------
template<int D>
static void run_gat(const float* X, const float* Efeat, const int* ei,
                    const float* Wl, const float* Wr, const float* We,
                    const float* att, const float* bias,
                    float* xl, float* xr, float* ea,
                    float* logitsB, unsigned* nmax, float* den,
                    float* out, int nN, int nE, int doRelu, hipStream_t s)
{
    // node transforms: keep L2-resident; edge transform: stream non-temporal
    L<128, D, false, false, false, false>(X, Wl, nullptr, xl, nN, s);
    L<128, D, false, false, false, false>(X, Wr, nullptr, xr, nN, s);
    L<128, D, true,  true,  false, false>(Efeat, We, nullptr, ea, nE, s);
    fill_u32<<<(nN + 255) / 256, 256, 0, s>>>(nmax, 0u, (size_t)nN);
    fill_u32<<<(nN + 255) / 256, 256, 0, s>>>((unsigned*)den, 0u, (size_t)nN);
    gat_logits<D><<<nE / 8, 256, 0, s>>>(xl, xr, ea, ei, att, logitsB, nmax, nE);
    gat_expden<<<(nE + 255) / 256, 256, 0, s>>>(logitsB, nmax, den, ei, nE);
    size_t tot = (size_t)nN * D;
    init_rows<<<(unsigned)((tot + 255) / 256), 256, 0, s>>>(out, bias, tot, D - 1);
    gat_scatter<D><<<nE / 8, 256, 0, s>>>(xl, logitsB, den, ei, out, nE);
    if (doRelu) relu_k<<<(unsigned)((tot + 255) / 256), 256, 0, s>>>(out, tot);
}

// --------------------------------- launch ---------------------------------
extern "C" void kernel_launch(void* const* d_in, const int* in_sizes, int n_in,
                              void* d_out, int out_size, void* d_ws, size_t ws_size,
                              hipStream_t stream)
{
    const int nN = in_sizes[0] / 32;   // NODE_F = 32
    const int nE = in_sizes[1] / 16;   // EDGE_F = 16
    const int G  = out_size - nN * 8;  // N_GRAPHS

    const float* x_nodes = (const float*)d_in[0];
    const float* e_raw   = (const float*)d_in[1];
    const int*   ei      = (const int*)d_in[2];
    const int*   batch   = (const int*)d_in[3];
    const float* Wn1 = (const float*)d_in[4],  *bn1 = (const float*)d_in[5];
    const float* Wn2 = (const float*)d_in[6],  *bn2 = (const float*)d_in[7];
    const float* We1 = (const float*)d_in[8],  *be1 = (const float*)d_in[9];
    const float* We2 = (const float*)d_in[10], *be2 = (const float*)d_in[11];
    const float* aWl = (const float*)d_in[12], *aWr = (const float*)d_in[13];
    const float* aWe = (const float*)d_in[14], *a_att = (const float*)d_in[15];
    const float* a_bias = (const float*)d_in[16];
    const float* c1Wl = (const float*)d_in[17], *c1Wr = (const float*)d_in[18];
    const float* c1We = (const float*)d_in[19], *c1_att = (const float*)d_in[20];
    const float* c1_bias = (const float*)d_in[21];
    const float* c2Wl = (const float*)d_in[22], *c2Wr = (const float*)d_in[23];
    const float* c2We = (const float*)d_in[24], *c2_att = (const float*)d_in[25];
    const float* c2_bias = (const float*)d_in[26];
    const float* Wc = (const float*)d_in[27], *bc = (const float*)d_in[28];

    // workspace carve (~540 MB): e_enc/ea dominate; node tensors sit in L2
    float* ws = (float*)d_ws;
    size_t off = 0;
    float* x_enc = ws + off; off += (size_t)nN * 128;
    float* e_enc = ws + off; off += (size_t)nE * 128;
    float* ea    = ws + off; off += (size_t)nE * 128;   // also encoder hidden
    float* xl    = ws + off; off += (size_t)nN * 128;
    float* xr    = ws + off; off += (size_t)nN * 128;
    float* h1    = ws + off; off += (size_t)nN * 128;
    float* h2    = ws + off; off += (size_t)nN * 128;
    float* logitsB = ws + off; off += (size_t)nE;
    unsigned* nmax = (unsigned*)(ws + off); off += (size_t)nN;
    float* den   = ws + off; off += (size_t)nN;
    float* pooled = ws + off; off += (size_t)G * 128;
    float* counts = ws + off; off += (size_t)G;
    float* hid = ea;  // encoder hidden aliases ea (dead before rounds start)

    float* actions = (float*)d_out;                   // [nN, 8]
    float* values  = (float*)d_out + (size_t)nN * 8;  // [G]

    // encoders (fc -> relu -> fc), all via WMMA; edge streams non-temporal
    L<32,  128, false, false, true, true >(x_nodes, Wn1, bn1, hid,   nN, stream);
    L<128, 128, false, false, true, false>(hid,     Wn2, bn2, x_enc, nN, stream);
    L<16,  128, true,  true,  true, true >(e_raw,   We1, be1, hid,   nE, stream);
    L<128, 128, true,  true,  true, false>(hid,     We2, be2, e_enc, nE, stream);

    // actor: one GATv2 round, D=8, straight into d_out
    run_gat<8>(x_enc, e_enc, ei, aWl, aWr, aWe, a_att, a_bias,
               xl, xr, ea, logitsB, nmax, den, actions, nN, nE, 0, stream);

    // critic: GATv2 -> relu -> GATv2
    run_gat<128>(x_enc, e_enc, ei, c1Wl, c1Wr, c1We, c1_att, c1_bias,
                 xl, xr, ea, logitsB, nmax, den, h1, nN, nE, 1, stream);
    run_gat<128>(h1, e_enc, ei, c2Wl, c2Wr, c2We, c2_att, c2_bias,
                 xl, xr, ea, logitsB, nmax, den, h2, nN, nE, 0, stream);

    // global mean pool + linear head
    fill_u32<<<(G * 128 + 255) / 256, 256, 0, stream>>>((unsigned*)pooled, 0u, (size_t)G * 128);
    fill_u32<<<1, 256, 0, stream>>>((unsigned*)counts, 0u, (size_t)G);
    pool_sum<<<(unsigned)(((size_t)nN * 128 + 255) / 256), 256, 0, stream>>>(h2, batch, pooled, nN);
    pool_cnt<<<(nN + 255) / 256, 256, 0, stream>>>(batch, counts, nN);
    graph_values<<<G, 32, 0, stream>>>(pooled, counts, Wc, bc, values, G);
}
